// Quantize_50302656971317
// MI455X (gfx1250) — matrix operand order repaired and checked by reading, not compiled, not run on hardware
//
#include <hip/hip_runtime.h>

typedef __bf16 bf16_t;
typedef __attribute__((ext_vector_type(16))) __bf16 v16bf;
typedef __attribute__((ext_vector_type(8)))  float  v8f;

#define DIM   256
#define NEMB  2048
#define TPB   512                 // 16 waves (wave32)
#define WAVES (TPB / 32)
#define ROWS_PER_BLOCK (WAVES * 16)   // 256 rows per block
#define BPAD  264                 // padded K-stride for B tiles in LDS (bank-conflict free)
#define NTILES (NEMB / 16)        // 128 column tiles

// ---------------------------------------------------------------------------
// Kernel 0: exact fp32 codebook column norms  ||e_j||^2  -> ws_norm[2048]
// ---------------------------------------------------------------------------
__global__ __launch_bounds__(256) void vq_norms(const float* __restrict__ embed,
                                                float* __restrict__ ws_norm) {
    int j = blockIdx.x * 256 + threadIdx.x;     // coalesced over j
    float s = 0.f;
    for (int k = 0; k < DIM; ++k) {
        float v = embed[(size_t)k * NEMB + j];
        s += v * v;
    }
    ws_norm[j] = s;
}

// ---------------------------------------------------------------------------
// Kernel 1: main VQ kernel.  Each wave owns a 16-row M-tile; the block sweeps
// all 2048 codes in 16-column tiles, double-buffered through LDS as bf16
// hi/lo.  Next tile's global loads are issued before the WMMA block so their
// latency hides under 24 v_wmma_f32_16x16x32_bf16 per wave per tile.  Three
// independent accumulators break the WMMA RAW chain (ISA 7.12.1: bf16
// WMMA->WMMA dep = 1 NOP + 4 co-exec slots) into 8-deep chains.
// ---------------------------------------------------------------------------
__global__ __launch_bounds__(TPB) void vq_main(const float* __restrict__ x,
                                               const float* __restrict__ embed,
                                               float* __restrict__ out_q,
                                               float* __restrict__ out_ind,
                                               const float* __restrict__ ws_norm,
                                               float* __restrict__ ws_partial) {
    __shared__ float  lds_en[NEMB];              // 8 KB   codebook norms
    __shared__ bf16_t lds_bh[2][16 * BPAD];      // 16.5 KB B tiles (hi), double buffered
    __shared__ bf16_t lds_bl[2][16 * BPAD];      // 16.5 KB B tiles (lo)
    __shared__ int    lds_j[ROWS_PER_BLOCK];     // winning code per row
    __shared__ float  lds_wdiff[WAVES];

    const int tid  = threadIdx.x;
    const int lane = tid & 31;
    const int warp = tid >> 5;
    const int hlf  = lane >> 4;              // 0: lanes 0-15, 1: lanes 16-31
    const int bc   = lane & 15;              // column within N-tile / row within M-tile
    const size_t rowBase = (size_t)blockIdx.x * ROWS_PER_BLOCK;

    // preload codebook norms into LDS (first tile barrier covers visibility)
    for (int j = tid; j < NEMB; j += TPB) lds_en[j] = ws_norm[j];

    // ---- load A fragments: 16 rows x 256 K as bf16 hi/lo, ISA A-layout ----
    // lanes 0-15 : elems 0-7 = K k0..k0+7,  elems 8-15 = K k0+16..k0+23 (k0 = 32*kc)
    // lanes 16-31: same with k0 = 32*kc + 8  -> K {8-15, 24-31}
    const int rowInBlk = warp * 16 + bc;
    const float* xrow = x + (rowBase + rowInBlk) * DIM;
    v16bf ah[8], al[8];
    float xn = 0.f;                           // exact fp32 row-norm partial
    #pragma unroll
    for (int kc = 0; kc < 8; ++kc) {
        const int k0 = kc * 32 + (hlf ? 8 : 0);
        float4 p0 = *(const float4*)(xrow + k0);
        float4 p1 = *(const float4*)(xrow + k0 + 4);
        float4 p2 = *(const float4*)(xrow + k0 + 16);
        float4 p3 = *(const float4*)(xrow + k0 + 20);
        float vals[16] = {p0.x,p0.y,p0.z,p0.w, p1.x,p1.y,p1.z,p1.w,
                          p2.x,p2.y,p2.z,p2.w, p3.x,p3.y,p3.z,p3.w};
        #pragma unroll
        for (int e = 0; e < 16; ++e) {
            float v = vals[e];
            xn += v * v;
            bf16_t h = (bf16_t)v;             // RTNE
            ah[kc][e] = h;
            al[kc][e] = (bf16_t)(v - (float)h);
        }
    }

    float bestv[8];
    int   besti[8];
    #pragma unroll
    for (int r = 0; r < 8; ++r) { bestv[r] = 3.4e38f; besti[r] = 0; }

    // ---- staging mapping: 4096 floats/tile = 1024 float4; 2 per thread ----
    // idx -> k = idx>>2 (0..255), seg = idx&3, cols n0 + seg*4 .. +3
    const int k0s  = tid >> 2,          seg0 = tid & 3;
    const int k1s  = (tid + TPB) >> 2,  seg1 = (tid + TPB) & 3;

    float4 pf0, pf1;                    // prefetch registers for next tile
    {   // prime the pipeline: tile 0
        pf0 = *(const float4*)&embed[(size_t)k0s * NEMB + 0 + seg0 * 4];
        pf1 = *(const float4*)&embed[(size_t)k1s * NEMB + 0 + seg1 * 4];
    }

    for (int tile = 0; tile < NTILES; ++tile) {
        const int n0  = tile * 16;
        const int buf = tile & 1;

        // convert + store the prefetched tile into LDS buffer `buf`
        {
            float f0[4] = {pf0.x, pf0.y, pf0.z, pf0.w};
            float f1[4] = {pf1.x, pf1.y, pf1.z, pf1.w};
            #pragma unroll
            for (int i = 0; i < 4; ++i) {
                int c0 = seg0 * 4 + i;
                bf16_t h0 = (bf16_t)f0[i];
                lds_bh[buf][c0 * BPAD + k0s] = h0;
                lds_bl[buf][c0 * BPAD + k0s] = (bf16_t)(f0[i] - (float)h0);
                int c1 = seg1 * 4 + i;
                bf16_t h1 = (bf16_t)f1[i];
                lds_bh[buf][c1 * BPAD + k1s] = h1;
                lds_bl[buf][c1 * BPAD + k1s] = (bf16_t)(f1[i] - (float)h1);
            }
        }
        __syncthreads();     // single barrier/tile: also proves tile-1 compute done

        // issue next tile's global loads now; latency hides under the WMMAs
        if (tile + 1 < NTILES) {
            const int nn = (tile + 1) * 16;
            pf0 = *(const float4*)&embed[(size_t)k0s * NEMB + nn + seg0 * 4];
            pf1 = *(const float4*)&embed[(size_t)k1s * NEMB + nn + seg1 * 4];
        }

        // three independent accumulator chains (hi*hi, lo*hi, hi*lo)
        v8f acc0 = {0.f,0.f,0.f,0.f,0.f,0.f,0.f,0.f};
        v8f acc1 = {0.f,0.f,0.f,0.f,0.f,0.f,0.f,0.f};
        v8f acc2 = {0.f,0.f,0.f,0.f,0.f,0.f,0.f,0.f};
        const int kbOff = hlf ? 8 : 0;       // B layout: lane holds K {kb..kb+7, kb+16..kb+23}
        #pragma unroll
        for (int kc = 0; kc < 8; ++kc) {
            const int kb = kc * 32 + kbOff;
            union { uint4 q[2]; v16bf v; } ub, ul;
            const uint4* ph = (const uint4*)&lds_bh[buf][bc * BPAD + kb];
            const uint4* pl = (const uint4*)&lds_bl[buf][bc * BPAD + kb];
            ub.q[0] = ph[0]; ub.q[1] = ph[2];
            ul.q[0] = pl[0]; ul.q[1] = pl[2];
            acc0 = __builtin_amdgcn_wmma_f32_16x16x32_bf16(false, ah[kc], false, ub.v,
                                                           (short)0, acc0, false, false);
            acc1 = __builtin_amdgcn_wmma_f32_16x16x32_bf16(false, al[kc], false, ub.v,
                                                           (short)0, acc1, false, false);
            acc2 = __builtin_amdgcn_wmma_f32_16x16x32_bf16(false, ah[kc], false, ul.v,
                                                           (short)0, acc2, false, false);
        }
        const float en  = lds_en[n0 + bc];
        const int   col = n0 + bc;
        #pragma unroll
        for (int r = 0; r < 8; ++r) {
            float dot = acc0[r] + acc1[r] + acc2[r];
            float d = en - 2.f * dot;        // ||e||^2 - 2 x.e  (x-norm is row-const)
            if (d < bestv[r]) { bestv[r] = d; besti[r] = col; }
        }
    }

    // ---- argmin reduce across the 16 lanes of each half (wave32 shuffles) ----
    #pragma unroll
    for (int r = 0; r < 8; ++r) {
        #pragma unroll
        for (int m = 1; m <= 8; m <<= 1) {
            float ov = __shfl_xor(bestv[r], m, 32);
            int   oi = __shfl_xor(besti[r], m, 32);
            if (ov < bestv[r] || (ov == bestv[r] && oi < besti[r])) {
                bestv[r] = ov; besti[r] = oi;
            }
        }
    }
    // exact row norm: lane m and m+16 hold complementary K halves of row m
    float xnTot = xn + __shfl_xor(xn, 16, 32);

    // diff per row = ||x||^2 + best_dist ; write indices + stash j for gather
    float dacc = 0.f;
    #pragma unroll
    for (int r = 0; r < 8; ++r) {
        float xr0 = __shfl(xnTot, r, 32);        // norm of row warp*16 + r
        float xr1 = __shfl(xnTot, r + 8, 32);    // norm of row warp*16 + 8 + r
        if (lane == 0) {                         // half0 result -> rows 0..7
            int row = warp * 16 + r;
            lds_j[row] = besti[r];
            out_ind[rowBase + row] = (float)besti[r];
            dacc += xr0 + bestv[r];
        }
        if (lane == 16) {                        // half1 result -> rows 8..15
            int row = warp * 16 + 8 + r;
            lds_j[row] = besti[r];
            out_ind[rowBase + row] = (float)besti[r];
            dacc += xr1 + bestv[r];
        }
    }
    dacc += __shfl_xor(dacc, 16, 32);
    if (lane == 0) lds_wdiff[warp] = dacc;
    __syncthreads();
    if (tid == 0) {
        float s = 0.f;
        for (int w = 0; w < WAVES; ++w) s += lds_wdiff[w];   // fixed order
        ws_partial[blockIdx.x] = s;
    }

    // ---- gather quantize rows: out_q[row,:] = embed[:, j_row] (L2-resident) ----
    for (int e = tid; e < ROWS_PER_BLOCK * DIM; e += TPB) {
        int row = e >> 8;                  // DIM == 256
        int k   = e & (DIM - 1);
        int j   = lds_j[row];
        out_q[(rowBase + row) * DIM + k] = embed[(size_t)k * NEMB + j];
    }
}

// ---------------------------------------------------------------------------
// Kernel 2: deterministic fixed-order reduction of per-block diff partials.
// ---------------------------------------------------------------------------
__global__ __launch_bounds__(256) void vq_reduce(const float* __restrict__ ws_partial,
                                                 float* __restrict__ out_diff,
                                                 int nblocks, float inv) {
    __shared__ float s[256];
    int t = threadIdx.x;
    float a = 0.f;
    for (int i = t; i < nblocks; i += 256) a += ws_partial[i];  // fixed order
    s[t] = a;
    __syncthreads();
    for (int off = 128; off > 0; off >>= 1) {
        if (t < off) s[t] += s[t + off];
        __syncthreads();
    }
    if (t == 0) out_diff[0] = s[0] * inv;
}

extern "C" void kernel_launch(void* const* d_in, const int* in_sizes, int n_in,
                              void* d_out, int out_size, void* d_ws, size_t ws_size,
                              hipStream_t stream) {
    const float* x     = (const float*)d_in[0];
    const float* embed = (const float*)d_in[1];
    const int nx    = in_sizes[0];            // 32*4096*256
    const int nrows = nx / DIM;               // 131072
    const int nblk  = nrows / ROWS_PER_BLOCK; // 512

    float* out      = (float*)d_out;
    float* out_q    = out;                    // [nrows, DIM]
    float* out_diff = out + (size_t)nx;       // scalar
    float* out_ind  = out + (size_t)nx + 1;   // [nrows] as float

    float* ws_norm    = (float*)d_ws;         // [NEMB]
    float* ws_partial = ws_norm + NEMB;       // [nblk]

    vq_norms<<<NEMB / 256, 256, 0, stream>>>(embed, ws_norm);
    vq_main<<<nblk, TPB, 0, stream>>>(x, embed, out_q, out_ind, ws_norm, ws_partial);
    vq_reduce<<<1, 256, 0, stream>>>(ws_partial, out_diff, nblk,
                                     1.0f / ((float)nrows * (float)DIM));
}
